// EncoderBlock_12120397709601
// MI455X (gfx1250) — matrix operand order repaired and checked by reading, not compiled, not run on hardware
//
#include <hip/hip_runtime.h>

#define D_  1024
#define H_  16
#define DK_ 64
#define FF_ 4096
#define S_  2048

typedef __attribute__((ext_vector_type(16))) __bf16       v16bf;
typedef __attribute__((ext_vector_type(8)))  float        v8f;
typedef __attribute__((ext_vector_type(8)))  unsigned int v8u;

__device__ __forceinline__ unsigned short f2bf(float f) {
    union { float f; unsigned u; } v; v.f = f;
    unsigned u = v.u;
    unsigned r = u + 0x7FFFu + ((u >> 16) & 1u);   // round to nearest even
    if ((u & 0x7F800000u) == 0x7F800000u) r = u;   // inf / nan: truncate
    return (unsigned short)(r >> 16);
}

// Build a 16xbf16 WMMA fragment from two 16B chunks (LDS or global).
__device__ __forceinline__ v16bf frag_from(const unsigned short* p0, const unsigned short* p1) {
    const uint4 lo = *(const uint4*)p0;
    const uint4 hi = *(const uint4*)p1;
    v8u raw = {lo.x, lo.y, lo.z, lo.w, hi.x, hi.y, hi.z, hi.w};
    return __builtin_bit_cast(v16bf, raw);
}

// --- CDNA5 async global->LDS DMA (ASYNCcnt-tracked). The instruction's
// offset immediate is added to BOTH the global and the LDS address, so one
// address pair covers a contiguous per-lane chunk.
__device__ __forceinline__ void async_copy_32B(const void* g, void* l) {
    unsigned lds = (unsigned)(size_t)l;   // low 32 bits of generic ptr = LDS offset
    asm volatile("global_load_async_to_lds_b128 %0, %1, off\n\t"
                 "global_load_async_to_lds_b128 %0, %1, off offset:16"
                 :: "v"(lds), "v"(g) : "memory");
}
__device__ __forceinline__ void async_copy_64B(const void* g, void* l) {
    unsigned lds = (unsigned)(size_t)l;
    asm volatile("global_load_async_to_lds_b128 %0, %1, off\n\t"
                 "global_load_async_to_lds_b128 %0, %1, off offset:16\n\t"
                 "global_load_async_to_lds_b128 %0, %1, off offset:32\n\t"
                 "global_load_async_to_lds_b128 %0, %1, off offset:48"
                 :: "v"(lds), "v"(g) : "memory");
}
__device__ __forceinline__ void wait_async0() {
    asm volatile("s_wait_asynccnt 0x0" ::: "memory");
}

// ---------------------------------------------------------------------------
// Generic bf16 WMMA GEMM: C[M,N] = A[M,K] * Bt[N,K]^T  (+bias, epilogue modes)
// block = 256 threads (8 waves), tile 128x128, k-step 32.
// Double-buffered LDS staged with global_load_async_to_lds_b128 so the next
// tile's DMA overlaps the current tile's WMMAs; one barrier per k-step.
// modes: 0=f32 store, 1=bf16 store, 2=relu+bf16, 3=Q/K head scatter, 4=V^T scatter
// ---------------------------------------------------------------------------
#define BM 128
#define BN 128
#define BK 32

__global__ __launch_bounds__(256) void gemm_bf16_kernel(
    const unsigned short* __restrict__ A, const unsigned short* __restrict__ Bt,
    const float* __restrict__ bias, void* __restrict__ Cout,
    int M, int N, int K, int mode)
{
    __shared__ unsigned short As[2][BM * BK];
    __shared__ unsigned short Bs[2][BN * BK];

    const int tid  = threadIdx.x;
    const int wave = tid >> 5, lane = tid & 31;
    const int hf   = lane >> 4, l15 = lane & 15;
    const int bm   = blockIdx.x * BM, bn = blockIdx.y * BN;
    const int wm   = (wave >> 1) * 32;   // 4 waves along M, 32 rows each
    const int wn   = (wave & 1) * 64;    // 2 waves along N, 64 cols each

    v8f acc[2][4] = {};

    const int ldrow = tid >> 1;          // 0..127
    const int ldcol = (tid & 1) * 16;    // 0 or 16 (16 bf16 = 32B per thread/tile)
    const unsigned short* Ag = A  + (size_t)(bm + ldrow) * K + ldcol;
    const unsigned short* Bg = Bt + (size_t)(bn + ldrow) * K + ldcol;
    const int soff = ldrow * BK + ldcol;

    const int nk = K / BK;
    // prologue: stage tile 0 (no LDS readers yet)
    async_copy_32B(Ag, &As[0][soff]);
    async_copy_32B(Bg, &Bs[0][soff]);

    for (int i = 0; i < nk; ++i) {
        const int cur = i & 1, nxt = cur ^ 1;
        wait_async0();        // current tile's DMA (issued last iteration) done
        __syncthreads();      // all waves: done reading buffer 'nxt', 'cur' visible
        if (i + 1 < nk) {     // prefetch next tile while we compute on 'cur'
            async_copy_32B(Ag + (i + 1) * BK, &As[nxt][soff]);
            async_copy_32B(Bg + (i + 1) * BK, &Bs[nxt][soff]);
        }

        v16bf af[2], bfr[4];
#pragma unroll
        for (int ii = 0; ii < 2; ii++) {
            const unsigned short* rp = &As[cur][(wm + ii * 16 + l15) * BK + hf * 8];
            af[ii] = frag_from(rp, rp + 16);
        }
#pragma unroll
        for (int j = 0; j < 4; j++) {
            const unsigned short* rp = &Bs[cur][(wn + j * 16 + l15) * BK + hf * 8];
            bfr[j] = frag_from(rp, rp + 16);
        }
#pragma unroll
        for (int ii = 0; ii < 2; ii++)
#pragma unroll
            for (int j = 0; j < 4; j++)
                acc[ii][j] = __builtin_amdgcn_wmma_f32_16x16x32_bf16(
                    false, af[ii], false, bfr[j], (short)0, acc[ii][j], false, false);
    }

    // epilogue
#pragma unroll
    for (int i = 0; i < 2; i++) {
#pragma unroll
        for (int j = 0; j < 4; j++) {
            const int col  = bn + wn + j * 16 + l15;
            const float bv = bias ? bias[col] : 0.0f;
#pragma unroll
            for (int r = 0; r < 8; r++) {
                const int row = bm + wm + i * 16 + hf * 8 + r;
                float v = acc[i][j][r] + bv;
                switch (mode) {
                case 0:
                    ((float*)Cout)[(size_t)row * N + col] = v; break;
                case 1:
                    ((unsigned short*)Cout)[(size_t)row * N + col] = f2bf(v); break;
                case 2:
                    ((unsigned short*)Cout)[(size_t)row * N + col] = f2bf(fmaxf(v, 0.0f)); break;
                case 3: {  // Q/K: [B,H,S,DK]
                    int bb = row / S_, s = row % S_, hh = col / DK_, dk = col % DK_;
                    ((unsigned short*)Cout)[(((size_t)bb * H_ + hh) * S_ + s) * DK_ + dk] = f2bf(v);
                } break;
                default: {  // 4: V^T: [B,H,DK,S]
                    int bb = row / S_, s = row % S_, hh = col / DK_, dk = col % DK_;
                    ((unsigned short*)Cout)[(((size_t)bb * H_ + hh) * DK_ + dk) * S_ + s] = f2bf(v);
                } break;
                }
            }
        }
    }
}

// ---------------------------------------------------------------------------
// Fused flash attention. grid = (S/128, H, B), block = 256 (8 waves).
// Q,K: bf16 [B,H,S,DK]; Vt: bf16 [B,H,DK,S]; out: bf16 [B*S, D] (head-merged)
// Each wave owns 16 q-rows; online softmax with width-16 shuffles.
// K/V tiles staged with async global->LDS DMA.
// ---------------------------------------------------------------------------
__global__ __launch_bounds__(256) void flash_attn_kernel(
    const unsigned short* __restrict__ Q, const unsigned short* __restrict__ Kg,
    const unsigned short* __restrict__ Vt, const unsigned char* __restrict__ mask,
    unsigned short* __restrict__ outp)
{
    __shared__ unsigned short KV[128 * 64];     // K tile [128][64]  OR  Vt tile [64][128]
    __shared__ unsigned short P[8 * 16 * 128];  // per-wave P staging (C-layout -> A-layout)

    const int tid  = threadIdx.x;
    const int wave = tid >> 5, lane = tid & 31;
    const int hf   = lane >> 4, l15 = lane & 15;
    const int qt   = blockIdx.x * 128;
    const int h    = blockIdx.y, b = blockIdx.z;

    const unsigned short* Qbh = Q  + ((size_t)(b * H_ + h)) * S_ * DK_;
    const unsigned short* Kbh = Kg + ((size_t)(b * H_ + h)) * S_ * DK_;
    const unsigned short* Vbh = Vt + ((size_t)(b * H_ + h)) * DK_ * S_;
    unsigned short* Pw = P + wave * 16 * 128;
    const int qw = qt + wave * 16;

    // Load this wave's Q fragments (16 rows x 64 d = 2 A-frags) straight from global.
    v16bf qf[2];
#pragma unroll
    for (int kk = 0; kk < 2; kk++) {
        const unsigned short* rp = Qbh + (size_t)(qw + l15) * DK_ + kk * 32 + hf * 8;
        qf[kk] = frag_from(rp, rp + 16);
    }

    v8f o[4] = {};
    float mprev[8], lsum[8];
#pragma unroll
    for (int r = 0; r < 8; r++) { mprev[r] = -__builtin_inff(); lsum[r] = 0.0f; }

    {   // per-thread staging addresses
    }

    for (int kb = 0; kb < S_; kb += 128) {
        __syncthreads();  // previous iteration done reading KV
        {   // async stage: K tile [128][64], 64B per thread
            int row = tid >> 1, c = (tid & 1) * 32;
            async_copy_64B(Kbh + (size_t)(kb + row) * DK_ + c, KV + row * 64 + c);
        }
        wait_async0();
        __syncthreads();

        // scores: 16 q-rows x 128 k-cols, K-dim = 64
        v8f sf[8] = {};
#pragma unroll
        for (int kk = 0; kk < 2; kk++) {
#pragma unroll
            for (int n = 0; n < 8; n++) {
                const unsigned short* rp = KV + (n * 16 + l15) * 64 + kk * 32 + hf * 8;
                v16bf bfr = frag_from(rp, rp + 16);
                sf[n] = __builtin_amdgcn_wmma_f32_16x16x32_bf16(
                    false, qf[kk], false, bfr, (short)0, sf[n], false, false);
            }
        }

        // online softmax (per register-row r; all 16 lanes of a half share a row)
#pragma unroll
        for (int r = 0; r < 8; r++) {
            const int row = qw + hf * 8 + r;
            const unsigned char* mp = mask + ((size_t)b * S_ + row) * S_ + kb + l15;
            float vals[8];
            float mx = -__builtin_inff();
#pragma unroll
            for (int n = 0; n < 8; n++) {
                float v = sf[n][r] * 0.125f;          // 1/sqrt(DK)
                if (mp[n * 16]) v = -__builtin_inff();
                vals[n] = v;
                mx = fmaxf(mx, v);
            }
#pragma unroll
            for (int off = 8; off; off >>= 1) mx = fmaxf(mx, __shfl_xor(mx, off, 16));
            const float mnew  = fmaxf(mprev[r], mx);
            const float alpha = (mnew == -__builtin_inff()) ? 1.0f : __expf(mprev[r] - mnew);
            mprev[r] = mnew;
            lsum[r] *= alpha;
#pragma unroll
            for (int d = 0; d < 4; d++) o[d][r] *= alpha;
            float rs = 0.0f;
#pragma unroll
            for (int n = 0; n < 8; n++) {
                float p = (mnew == -__builtin_inff()) ? 0.0f : __expf(vals[n] - mnew);
                rs += p;
                Pw[(hf * 8 + r) * 128 + n * 16 + l15] = f2bf(p);
            }
#pragma unroll
            for (int off = 8; off; off >>= 1) rs += __shfl_xor(rs, off, 16);
            lsum[r] += rs;
        }

        __syncthreads();  // all waves done reading K tile
        {   // async stage: Vt tile [64][128], 64B per thread
            int row = tid >> 2, c = (tid & 3) * 32;
            async_copy_64B(Vbh + (size_t)row * S_ + kb + c, KV + row * 128 + c);
        }
        wait_async0();
        __syncthreads();

        // O += P (16x128) @ V (128x64)
#pragma unroll
        for (int kk = 0; kk < 4; kk++) {
            const unsigned short* pr = Pw + l15 * 128 + kk * 32 + hf * 8;
            v16bf pa = frag_from(pr, pr + 16);
#pragma unroll
            for (int d = 0; d < 4; d++) {
                const unsigned short* rp = KV + (d * 16 + l15) * 128 + kk * 32 + hf * 8;
                v16bf vb = frag_from(rp, rp + 16);
                o[d] = __builtin_amdgcn_wmma_f32_16x16x32_bf16(
                    false, pa, false, vb, (short)0, o[d], false, false);
            }
        }
    }

    // finalize: divide by row-sum (0 for fully-masked rows) and write head-merged bf16
#pragma unroll
    for (int r = 0; r < 8; r++) {
        const float inv = (lsum[r] > 0.0f) ? 1.0f / lsum[r] : 0.0f;
        const int row = qw + hf * 8 + r;
#pragma unroll
        for (int d = 0; d < 4; d++) {
            const int col = h * DK_ + d * 16 + l15;
            outp[((size_t)b * S_ + row) * D_ + col] = f2bf(o[d][r] * inv);
        }
    }
}

// ---------------------------------------------------------------------------
// out = LayerNorm(a + res) * gamma + beta ; writes f32 (+optional bf16 copy)
// one 256-thread block per row of D_=1024.
// ---------------------------------------------------------------------------
__global__ __launch_bounds__(256) void ln_res_kernel(
    const float* __restrict__ a, const float* __restrict__ res,
    const float* __restrict__ gamma, const float* __restrict__ beta,
    float* __restrict__ outf, unsigned short* __restrict__ outb)
{
    __shared__ float red1[8], red2[8];
    const int row = blockIdx.x, tid = threadIdx.x;
    const int wave = tid >> 5, lane = tid & 31;
    const size_t base = (size_t)row * D_;

    float v[4]; float s = 0.0f, s2 = 0.0f;
#pragma unroll
    for (int i = 0; i < 4; i++) {
        const int c = tid + i * 256;
        const float x = a[base + c] + res[base + c];
        v[i] = x; s += x; s2 += x * x;
    }
#pragma unroll
    for (int off = 16; off; off >>= 1) { s += __shfl_xor(s, off, 32); s2 += __shfl_xor(s2, off, 32); }
    if (lane == 0) { red1[wave] = s; red2[wave] = s2; }
    __syncthreads();
    float ts = 0.0f, t2 = 0.0f;
#pragma unroll
    for (int w = 0; w < 8; w++) { ts += red1[w]; t2 += red2[w]; }
    const float mu   = ts * (1.0f / D_);
    const float var  = t2 * (1.0f / D_) - mu * mu;
    const float rstd = rsqrtf(var + 1e-5f);
#pragma unroll
    for (int i = 0; i < 4; i++) {
        const int c = tid + i * 256;
        const float y = (v[i] - mu) * rstd * gamma[c] + beta[c];
        outf[base + c] = y;
        if (outb) outb[base + c] = f2bf(y);
    }
}

__global__ void cvt_bf16_kernel(const float* __restrict__ in, unsigned short* __restrict__ out, int n) {
    const int i = blockIdx.x * 256 + threadIdx.x;
    if (i < n) out[i] = f2bf(in[i]);
}

// in: f32 [R,C] row-major -> out: bf16 [C,R] row-major (weight transpose+convert)
__global__ void transpose_cvt_kernel(const float* __restrict__ in, unsigned short* __restrict__ out,
                                     int R, int C) {
    const int i = blockIdx.x * 256 + threadIdx.x;
    if (i < R * C) {
        const int r = i / C, c = i % C;
        out[(size_t)c * R + r] = f2bf(in[i]);
    }
}

// ---------------------------------------------------------------------------
extern "C" void kernel_launch(void* const* d_in, const int* in_sizes, int n_in,
                              void* d_out, int out_size, void* d_ws, size_t ws_size,
                              hipStream_t stream)
{
    (void)n_in; (void)out_size; (void)ws_size;
    const float*         x    = (const float*)d_in[0];
    const unsigned char* mask = (const unsigned char*)d_in[1];
    const float* Wq  = (const float*)d_in[2];
    const float* Wk  = (const float*)d_in[3];
    const float* Wv  = (const float*)d_in[4];
    const float* Wo  = (const float*)d_in[5];
    const float* bo  = (const float*)d_in[6];
    const float* W1  = (const float*)d_in[7];
    const float* b1  = (const float*)d_in[8];
    const float* W2  = (const float*)d_in[9];
    const float* b2  = (const float*)d_in[10];
    const float* g1p = (const float*)d_in[11];
    const float* be1 = (const float*)d_in[12];
    const float* g2p = (const float*)d_in[13];
    const float* be2 = (const float*)d_in[14];

    const int M  = in_sizes[0] / D_;   // B*S = 4096
    const int Bb = M / S_;             // batch = 2

    char* ws = (char*)d_ws;
    size_t off = 0;
    auto alloc = [&](size_t bytes) -> char* {
        char* p = ws + off;
        off += (bytes + 255) & ~(size_t)255;
        return p;
    };
    unsigned short* xb   = (unsigned short*)alloc((size_t)M * D_ * 2);
    unsigned short* wqT  = (unsigned short*)alloc((size_t)D_ * D_ * 2);
    unsigned short* wkT  = (unsigned short*)alloc((size_t)D_ * D_ * 2);
    unsigned short* wvT  = (unsigned short*)alloc((size_t)D_ * D_ * 2);
    unsigned short* woT  = (unsigned short*)alloc((size_t)D_ * D_ * 2);
    unsigned short* w1T  = (unsigned short*)alloc((size_t)D_ * FF_ * 2);
    unsigned short* w2T  = (unsigned short*)alloc((size_t)FF_ * D_ * 2);
    unsigned short* Qh   = (unsigned short*)alloc((size_t)M * D_ * 2);
    unsigned short* Kh   = (unsigned short*)alloc((size_t)M * D_ * 2);
    unsigned short* VhT  = (unsigned short*)alloc((size_t)M * D_ * 2);
    unsigned short* aout = (unsigned short*)alloc((size_t)M * D_ * 2);
    float*          proj = (float*)alloc((size_t)M * D_ * 4);
    float*          x1   = (float*)alloc((size_t)M * D_ * 4);
    unsigned short* x1b  = (unsigned short*)alloc((size_t)M * D_ * 2);
    unsigned short* h1b  = (unsigned short*)alloc((size_t)M * FF_ * 2);
    float*          ffo  = proj;  // reuse: proj is dead after first LayerNorm

    const dim3 blk(256);
    {
        int n = M * D_;
        cvt_bf16_kernel<<<dim3((n + 255) / 256), blk, 0, stream>>>(x, xb, n);
    }
    {
        int n = D_ * D_;
        dim3 g((n + 255) / 256);
        transpose_cvt_kernel<<<g, blk, 0, stream>>>(Wq, wqT, D_, D_);
        transpose_cvt_kernel<<<g, blk, 0, stream>>>(Wk, wkT, D_, D_);
        transpose_cvt_kernel<<<g, blk, 0, stream>>>(Wv, wvT, D_, D_);
        transpose_cvt_kernel<<<g, blk, 0, stream>>>(Wo, woT, D_, D_);
        int n2 = D_ * FF_;
        transpose_cvt_kernel<<<dim3((n2 + 255) / 256), blk, 0, stream>>>(W1, w1T, D_, FF_);
        transpose_cvt_kernel<<<dim3((n2 + 255) / 256), blk, 0, stream>>>(W2, w2T, FF_, D_);
    }

    const dim3 gDD(M / BM, D_ / BN);
    gemm_bf16_kernel<<<gDD, blk, 0, stream>>>(xb, wqT, nullptr, Qh,  M, D_, D_, 3);
    gemm_bf16_kernel<<<gDD, blk, 0, stream>>>(xb, wkT, nullptr, Kh,  M, D_, D_, 3);
    gemm_bf16_kernel<<<gDD, blk, 0, stream>>>(xb, wvT, nullptr, VhT, M, D_, D_, 4);

    flash_attn_kernel<<<dim3(S_ / 128, H_, Bb), blk, 0, stream>>>(Qh, Kh, VhT, mask, aout);

    gemm_bf16_kernel<<<gDD, blk, 0, stream>>>(aout, woT, bo, proj, M, D_, D_, 0);
    ln_res_kernel<<<dim3(M), blk, 0, stream>>>(proj, x, g1p, be1, x1, x1b);

    gemm_bf16_kernel<<<dim3(M / BM, FF_ / BN), blk, 0, stream>>>(x1b, w1T, b1, h1b, M, FF_, D_, 2);
    gemm_bf16_kernel<<<gDD, blk, 0, stream>>>(h1b, w2T, b2, ffo, M, D_, FF_, 0);
    ln_res_kernel<<<dim3(M), blk, 0, stream>>>(ffo, x1, g2p, be2, (float*)d_out, nullptr);
}
